// ConvAttention_71975061946811
// MI455X (gfx1250) — compile-verified, matrix operand order checked by reading, not run on hardware
//
#include <hip/hip_runtime.h>

typedef __attribute__((ext_vector_type(16))) _Float16 v16h;
typedef __attribute__((ext_vector_type(8)))  _Float16 v8h;
typedef __attribute__((ext_vector_type(8)))  float    v8f;

#define TOK 4096      // B*L
#define DIM 1024
#define DFF 4096

static __device__ __forceinline__ v16h mk16(v8h lo, v8h hi) {
  v16h r;
#pragma unroll
  for (int i = 0; i < 8; ++i) { r[i] = lo[i]; r[i + 8] = hi[i]; }
  return r;
}

static __device__ __forceinline__ v8f wmma_f16(v16h a, v16h b, v8f c) {
  return __builtin_amdgcn_wmma_f32_16x16x32_f16(false, a, false, b, (short)0, c, false, false);
}

// ---------------------------------------------------------------- convert
__global__ void f32_to_f16_kernel(const float* __restrict__ in,
                                  _Float16* __restrict__ out, long n) {
  long i = (long)blockIdx.x * blockDim.x + threadIdx.x;
  long stride = (long)gridDim.x * blockDim.x;
  for (; i < n; i += stride) out[i] = (_Float16)in[i];
}

// ---------------------------------------------------------------- GEMM
// C[M,N] = A[M,K] (f16 row-major) dot B[N,K] (f16 row-major, i.e. X @ W^T)
// Ping-pong LDS double buffering: global loads for chunk k+1 issue before the
// WMMAs of chunk k, LDS stores after, single barrier per iteration.
enum { EPI_BIAS_H = 0, EPI_BIAS_F = 1, EPI_BN_GELU_H = 2, EPI_BN_F = 3 };

template <int EPI>
__global__ __launch_bounds__(256) void gemm_wmma(
    const _Float16* __restrict__ A, const _Float16* __restrict__ B,
    const float* __restrict__ bias,
    const float* __restrict__ bng, const float* __restrict__ bnb,
    const float* __restrict__ bnm, const float* __restrict__ bnv,
    float* __restrict__ Cf, _Float16* __restrict__ Ch,
    int M, int N, int K) {
  __shared__ __align__(16) _Float16 As[2][128][40];
  __shared__ __align__(16) _Float16 Bs[2][128][40];

  const int tid  = threadIdx.x;
  const int lane = tid & 31;
  const int wave = tid >> 5;          // 8 waves: 4 along M, 2 along N
  const int wm = wave & 3, wn = wave >> 2;
  const int bm = blockIdx.y * 128, bn0 = blockIdx.x * 128;
  const int lr = lane & 15;           // lane row/col within tile
  const int lh = lane >> 4;           // lane half selects K sub-pattern
  const int ksel = lh ? 8 : 0;

  // loader mapping: 512 v8h chunks per tile, 2 per thread
  int lrow[2], lcol[2];
#pragma unroll
  for (int j = 0; j < 2; ++j) {
    int c = tid + j * 256;
    lrow[j] = c >> 2;
    lcol[j] = (c & 3) << 3;
  }

  v8f acc[2][4];
#pragma unroll
  for (int mt = 0; mt < 2; ++mt)
#pragma unroll
    for (int nt = 0; nt < 4; ++nt) acc[mt][nt] = (v8f){};

  // prologue: chunk 0 -> buffer 0
#pragma unroll
  for (int j = 0; j < 2; ++j) {
    *(v8h*)&As[0][lrow[j]][lcol[j]] =
        *(const v8h*)&A[(size_t)(bm + lrow[j]) * K + lcol[j]];
    *(v8h*)&Bs[0][lrow[j]][lcol[j]] =
        *(const v8h*)&B[(size_t)(bn0 + lrow[j]) * K + lcol[j]];
  }
  __syncthreads();

  int cur = 0;
  for (int k0 = 0; k0 < K; k0 += 32) {
    const bool hasNext = (k0 + 32 < K);
    v8h pa[2], pb[2];
    if (hasNext) {
#pragma unroll
      for (int j = 0; j < 2; ++j) {
        pa[j] = *(const v8h*)&A[(size_t)(bm + lrow[j]) * K + k0 + 32 + lcol[j]];
        pb[j] = *(const v8h*)&B[(size_t)(bn0 + lrow[j]) * K + k0 + 32 + lcol[j]];
      }
    }

    v16h af[2], bf[4];
#pragma unroll
    for (int mt = 0; mt < 2; ++mt) {
      int r = wm * 32 + mt * 16 + lr;
      af[mt] = mk16(*(const v8h*)&As[cur][r][ksel],
                    *(const v8h*)&As[cur][r][ksel + 16]);
    }
#pragma unroll
    for (int nt = 0; nt < 4; ++nt) {
      int c = wn * 64 + nt * 16 + lr;
      bf[nt] = mk16(*(const v8h*)&Bs[cur][c][ksel],
                    *(const v8h*)&Bs[cur][c][ksel + 16]);
    }
#pragma unroll
    for (int mt = 0; mt < 2; ++mt)
#pragma unroll
      for (int nt = 0; nt < 4; ++nt)
        acc[mt][nt] = wmma_f16(af[mt], bf[nt], acc[mt][nt]);

    if (hasNext) {
#pragma unroll
      for (int j = 0; j < 2; ++j) {
        *(v8h*)&As[cur ^ 1][lrow[j]][lcol[j]] = pa[j];
        *(v8h*)&Bs[cur ^ 1][lrow[j]][lcol[j]] = pb[j];
      }
    }
    __syncthreads();
    cur ^= 1;
  }

  // epilogue: per-column params hoisted out of the row loops
#pragma unroll
  for (int nt = 0; nt < 4; ++nt) {
    const int col = bn0 + wn * 64 + nt * 16 + lr;
    float cadd = 0.f, cscale = 1.f, cshift = 0.f;
    if (EPI == EPI_BIAS_H || EPI == EPI_BIAS_F) {
      cadd = bias[col];
    } else {
      cscale = bng[col] * rsqrtf(bnv[col] + 1e-5f);
      cshift = bnb[col] - bnm[col] * cscale;
    }
#pragma unroll
    for (int mt = 0; mt < 2; ++mt)
#pragma unroll
      for (int v = 0; v < 8; ++v) {
        const int row = bm + wm * 32 + mt * 16 + v + (lh ? 8 : 0);
        float x = acc[mt][nt][v];
        if (EPI == EPI_BIAS_H) {
          Ch[(size_t)row * N + col] = (_Float16)(x + cadd);
        } else if (EPI == EPI_BIAS_F) {
          Cf[(size_t)row * N + col] = x + cadd;
        } else if (EPI == EPI_BN_GELU_H) {
          x = x * cscale + cshift;
          x = 0.5f * x * (1.0f + erff(x * 0.70710678118654752f));
          Ch[(size_t)row * N + col] = (_Float16)x;
        } else {
          Cf[(size_t)row * N + col] = x * cscale + cshift;
        }
      }
  }
}

// ---------------------------------------------------------------- attention
// One block = one (b, h, 64-row q tile); 4 waves x 16 q-rows each.
// Online softmax over 32-key chunks. K kept key-major (contiguous hd) for
// S = Q.K^T; V transposed into Vt[hd][key] at load time so the P.V B-operand
// is two contiguous ds_load_b128 per 16x16 tile. Ping-pong K/V buffers.
__global__ __launch_bounds__(128) void attn_flash(
    const _Float16* __restrict__ Q, const _Float16* __restrict__ Km,
    const _Float16* __restrict__ V, _Float16* __restrict__ O) {
  __shared__ __align__(16) _Float16 Qs[64][72];
  __shared__ __align__(16) _Float16 Ks[2][32][72];
  __shared__ __align__(16) _Float16 Vt[2][64][40];
  __shared__ __align__(16) _Float16 Ps[4][16][40];

  const int tid  = threadIdx.x;
  const int lane = tid & 31;
  const int wave = tid >> 5;
  const int lr = lane & 15, lh = lane >> 4, ksel = lh ? 8 : 0;

  const int bz = blockIdx.x;
  const int qt = bz & 15;            // q tile (64 rows)
  const int h  = (bz >> 4) & 15;
  const int b  = bz >> 8;
  const size_t base = (size_t)b * 1024 * DIM + (size_t)h * 64;

  // loader mapping for K/V chunks: 256 v8h chunks, 2 per thread
  int crow[2], ccol[2];
#pragma unroll
  for (int u = 0; u < 2; ++u) {
    int c = tid + u * 128;
    crow[u] = c >> 3;
    ccol[u] = (c & 7) << 3;
  }

  // Q tile 64x64 (512 chunks, 4 per thread)
  for (int c = tid; c < 64 * 8; c += 128) {
    int row = c >> 3, col = (c & 7) << 3;
    *(v8h*)&Qs[row][col] =
        *(const v8h*)&Q[base + (size_t)(qt * 64 + row) * DIM + col];
  }
  // prologue: chunk 0 of K/V -> buffer 0 (V transposed)
#pragma unroll
  for (int u = 0; u < 2; ++u) {
    *(v8h*)&Ks[0][crow[u]][ccol[u]] =
        *(const v8h*)&Km[base + (size_t)crow[u] * DIM + ccol[u]];
    v8h vv = *(const v8h*)&V[base + (size_t)crow[u] * DIM + ccol[u]];
#pragma unroll
    for (int i = 0; i < 8; ++i) Vt[0][ccol[u] + i][crow[u]] = vv[i];
  }
  __syncthreads();

  v16h qf[2];
  {
    int r = wave * 16 + lr;
    qf[0] = mk16(*(const v8h*)&Qs[r][ksel], *(const v8h*)&Qs[r][ksel + 16]);
    qf[1] = mk16(*(const v8h*)&Qs[r][32 + ksel], *(const v8h*)&Qs[r][32 + ksel + 16]);
  }

  float m[8], l[8];
  v8f acc[4];
#pragma unroll
  for (int v = 0; v < 8; ++v) { m[v] = -1e30f; l[v] = 0.f; }
#pragma unroll
  for (int nt = 0; nt < 4; ++nt) acc[nt] = (v8f){};

  const float scale = 0.125f;   // hd^-0.5, hd=64

  int cur = 0;
  for (int j = 0; j < 1024; j += 32) {
    const bool hasNext = (j + 32 < 1024);
    v8h kn[2], vn[2];
    if (hasNext) {
#pragma unroll
      for (int u = 0; u < 2; ++u) {
        kn[u] = *(const v8h*)&Km[base + (size_t)(j + 32 + crow[u]) * DIM + ccol[u]];
        vn[u] = *(const v8h*)&V [base + (size_t)(j + 32 + crow[u]) * DIM + ccol[u]];
      }
    }

    // S = Q (16x64) . K^T (64x32) -> two 16x16 f32 tiles
    v8f s[2];
#pragma unroll
    for (int st = 0; st < 2; ++st) {
      int kr = st * 16 + lr;
      v16h b0 = mk16(*(const v8h*)&Ks[cur][kr][ksel],
                     *(const v8h*)&Ks[cur][kr][ksel + 16]);
      v16h b1 = mk16(*(const v8h*)&Ks[cur][kr][32 + ksel],
                     *(const v8h*)&Ks[cur][kr][32 + ksel + 16]);
      v8f z = (v8f){};
      z = wmma_f16(qf[0], b0, z);
      z = wmma_f16(qf[1], b1, z);
      s[st] = z;
    }

    // online softmax row update (rows live in 16-lane groups; xor<16 stays in group)
    float corr[8];
#pragma unroll
    for (int v = 0; v < 8; ++v) {
      s[0][v] *= scale; s[1][v] *= scale;
      float t = fmaxf(s[0][v], s[1][v]);
      t = fmaxf(t, __shfl_xor(t, 1, 32));
      t = fmaxf(t, __shfl_xor(t, 2, 32));
      t = fmaxf(t, __shfl_xor(t, 4, 32));
      t = fmaxf(t, __shfl_xor(t, 8, 32));
      float mn = fmaxf(m[v], t);
      corr[v] = expf(m[v] - mn);
      s[0][v] = expf(s[0][v] - mn);
      s[1][v] = expf(s[1][v] - mn);
      float r = s[0][v] + s[1][v];
      r += __shfl_xor(r, 1, 32);
      r += __shfl_xor(r, 2, 32);
      r += __shfl_xor(r, 4, 32);
      r += __shfl_xor(r, 8, 32);
      l[v] = l[v] * corr[v] + r;
      m[v] = mn;
    }

    // stage P (C layout -> A layout) through per-wave LDS; same-wave DS ops are
    // in-order, a DScnt drain is enough (no block barrier needed).
#pragma unroll
    for (int v = 0; v < 8; ++v) {
      int r = v + (lh ? 8 : 0);
      Ps[wave][r][lr]      = (_Float16)s[0][v];
      Ps[wave][r][16 + lr] = (_Float16)s[1][v];
    }
    asm volatile("s_wait_dscnt 0x0" ::: "memory");
    v16h pf = mk16(*(const v8h*)&Ps[wave][lr][ksel],
                   *(const v8h*)&Ps[wave][lr][ksel + 16]);

    // acc = acc*corr + P (16x32) . V (32x64); B operand contiguous from Vt
#pragma unroll
    for (int nt = 0; nt < 4; ++nt) {
#pragma unroll
      for (int v = 0; v < 8; ++v) acc[nt][v] *= corr[v];
      v16h bvf = mk16(*(const v8h*)&Vt[cur][nt * 16 + lr][ksel],
                      *(const v8h*)&Vt[cur][nt * 16 + lr][ksel + 16]);
      acc[nt] = wmma_f16(pf, bvf, acc[nt]);
    }

    if (hasNext) {
#pragma unroll
      for (int u = 0; u < 2; ++u) {
        *(v8h*)&Ks[cur ^ 1][crow[u]][ccol[u]] = kn[u];
#pragma unroll
        for (int i = 0; i < 8; ++i) Vt[cur ^ 1][ccol[u] + i][crow[u]] = vn[u][i];
      }
    }
    __syncthreads();
    cur ^= 1;
  }

#pragma unroll
  for (int nt = 0; nt < 4; ++nt)
#pragma unroll
    for (int v = 0; v < 8; ++v) {
      int row = qt * 64 + wave * 16 + v + (lh ? 8 : 0);
      int col = nt * 16 + lr;
      O[base + (size_t)row * DIM + col] = (_Float16)(acc[nt][v] / l[v]);
    }
}

// ---------------------------------------------------------------- residual + LN
__global__ __launch_bounds__(256) void residual_ln(
    const float* __restrict__ X, const float* __restrict__ R,
    const float* __restrict__ g, const float* __restrict__ bta,
    float* __restrict__ Of, _Float16* __restrict__ Oh) {
  __shared__ float red[16];
  const size_t base = (size_t)blockIdx.x * DIM;
  const int i0 = threadIdx.x * 4;
  float v[4];
  float s = 0.f, s2 = 0.f;
#pragma unroll
  for (int i = 0; i < 4; ++i) {
    float x = X[base + i0 + i] + R[base + i0 + i];
    v[i] = x; s += x; s2 += x * x;
  }
  for (int msk = 1; msk < 32; msk <<= 1) {
    s  += __shfl_xor(s,  msk, 32);
    s2 += __shfl_xor(s2, msk, 32);
  }
  const int wave = threadIdx.x >> 5, lane = threadIdx.x & 31;
  if (lane == 0) { red[wave] = s; red[8 + wave] = s2; }
  __syncthreads();
  if (wave == 0) {
    float a  = (lane < 8) ? red[lane] : 0.f;
    float a2 = (lane < 8) ? red[8 + lane] : 0.f;
    for (int msk = 1; msk < 8; msk <<= 1) {
      a  += __shfl_xor(a,  msk, 32);
      a2 += __shfl_xor(a2, msk, 32);
    }
    if (lane == 0) { red[0] = a; red[1] = a2; }
  }
  __syncthreads();
  const float mean = red[0] * (1.f / DIM);
  const float var  = red[1] * (1.f / DIM) - mean * mean;
  const float inv  = rsqrtf(var + 1e-5f);
#pragma unroll
  for (int i = 0; i < 4; ++i) {
    float y = (v[i] - mean) * inv * g[i0 + i] + bta[i0 + i];
    Of[base + i0 + i] = y;
    if (Oh) Oh[base + i0 + i] = (_Float16)y;
  }
}

// ---------------------------------------------------------------- launch
extern "C" void kernel_launch(void* const* d_in, const int* in_sizes, int n_in,
                              void* d_out, int out_size, void* d_ws, size_t ws_size,
                              hipStream_t stream) {
  (void)in_sizes; (void)n_in; (void)out_size; (void)ws_size;
  const float* q     = (const float*)d_in[0];
  const float* wq    = (const float*)d_in[1];
  const float* bq    = (const float*)d_in[2];
  const float* wk    = (const float*)d_in[3];
  const float* bk    = (const float*)d_in[4];
  const float* wv    = (const float*)d_in[5];
  const float* bv    = (const float*)d_in[6];
  const float* wo    = (const float*)d_in[7];
  const float* bo    = (const float*)d_in[8];
  const float* w1    = (const float*)d_in[9];
  const float* bn1_g = (const float*)d_in[10];
  const float* bn1_b = (const float*)d_in[11];
  const float* bn1_m = (const float*)d_in[12];
  const float* bn1_v = (const float*)d_in[13];
  const float* w2    = (const float*)d_in[14];
  const float* bn2_g = (const float*)d_in[15];
  const float* bn2_b = (const float*)d_in[16];
  const float* bn2_m = (const float*)d_in[17];
  const float* bn2_v = (const float*)d_in[18];
  const float* ln1_g = (const float*)d_in[19];
  const float* ln1_b = (const float*)d_in[20];
  const float* ln3_g = (const float*)d_in[21];
  const float* ln3_b = (const float*)d_in[22];

  char* ws = (char*)d_ws;
  size_t off = 0;
  auto alloc = [&](size_t bytes) -> void* {
    void* p = ws + off;
    off += (bytes + 255) & ~(size_t)255;
    return p;
  };

  _Float16* Xh  = (_Float16*)alloc((size_t)TOK * DIM * 2);
  _Float16* Wqh = (_Float16*)alloc((size_t)DIM * DIM * 2);
  _Float16* Wkh = (_Float16*)alloc((size_t)DIM * DIM * 2);
  _Float16* Wvh = (_Float16*)alloc((size_t)DIM * DIM * 2);
  _Float16* Woh = (_Float16*)alloc((size_t)DIM * DIM * 2);
  _Float16* W1h = (_Float16*)alloc((size_t)DFF * DIM * 2);
  _Float16* W2h = (_Float16*)alloc((size_t)DIM * DFF * 2);
  _Float16* Qh  = (_Float16*)alloc((size_t)TOK * DIM * 2);
  _Float16* Kh  = (_Float16*)alloc((size_t)TOK * DIM * 2);
  _Float16* Vh  = (_Float16*)alloc((size_t)TOK * DIM * 2);
  _Float16* Ah  = (_Float16*)alloc((size_t)TOK * DIM * 2);
  float*    AO  = (float*)   alloc((size_t)TOK * DIM * 4);
  float*    Xf  = (float*)   alloc((size_t)TOK * DIM * 4);
  _Float16* Xh2 = (_Float16*)alloc((size_t)TOK * DIM * 2);
  _Float16* Hh  = (_Float16*)alloc((size_t)TOK * DFF * 2);
  float*    H2  = (float*)   alloc((size_t)TOK * DIM * 4);

  f32_to_f16_kernel<<<1024, 256, 0, stream>>>(q,  Xh,  (long)TOK * DIM);
  f32_to_f16_kernel<<<512,  256, 0, stream>>>(wq, Wqh, (long)DIM * DIM);
  f32_to_f16_kernel<<<512,  256, 0, stream>>>(wk, Wkh, (long)DIM * DIM);
  f32_to_f16_kernel<<<512,  256, 0, stream>>>(wv, Wvh, (long)DIM * DIM);
  f32_to_f16_kernel<<<512,  256, 0, stream>>>(wo, Woh, (long)DIM * DIM);
  f32_to_f16_kernel<<<1024, 256, 0, stream>>>(w1, W1h, (long)DFF * DIM);
  f32_to_f16_kernel<<<1024, 256, 0, stream>>>(w2, W2h, (long)DIM * DFF);

  dim3 gD(DIM / 128, TOK / 128);   // (8, 32)
  dim3 gF(DFF / 128, TOK / 128);   // (32, 32)

  gemm_wmma<EPI_BIAS_H><<<gD, 256, 0, stream>>>(
      Xh, Wqh, bq, nullptr, nullptr, nullptr, nullptr, nullptr, Qh, TOK, DIM, DIM);
  gemm_wmma<EPI_BIAS_H><<<gD, 256, 0, stream>>>(
      Xh, Wkh, bk, nullptr, nullptr, nullptr, nullptr, nullptr, Kh, TOK, DIM, DIM);
  gemm_wmma<EPI_BIAS_H><<<gD, 256, 0, stream>>>(
      Xh, Wvh, bv, nullptr, nullptr, nullptr, nullptr, nullptr, Vh, TOK, DIM, DIM);

  attn_flash<<<1024, 128, 0, stream>>>(Qh, Kh, Vh, Ah);

  gemm_wmma<EPI_BIAS_F><<<gD, 256, 0, stream>>>(
      Ah, Woh, bo, nullptr, nullptr, nullptr, nullptr, AO, nullptr, TOK, DIM, DIM);

  residual_ln<<<TOK, 256, 0, stream>>>(q, AO, ln1_g, ln1_b, Xf, Xh2);

  gemm_wmma<EPI_BN_GELU_H><<<gF, 256, 0, stream>>>(
      Xh2, W1h, nullptr, bn1_g, bn1_b, bn1_m, bn1_v, nullptr, Hh, TOK, DFF, DIM);

  gemm_wmma<EPI_BN_F><<<gD, 256, 0, stream>>>(
      Hh, W2h, nullptr, bn2_g, bn2_b, bn2_m, bn2_v, H2, nullptr, TOK, DIM, DFF);

  residual_ln<<<TOK, 256, 0, stream>>>(Xf, H2, ln3_g, ln3_b, (float*)d_out, nullptr);
}
